// NeighborhoodSelfAttention_23776938951308
// MI455X (gfx1250) — matrix operand order repaired and checked by reading, not compiled, not run on hardware
//
#include <hip/hip_runtime.h>
#include <hip/hip_bf16.h>

// ---------------------------------------------------------------------------
// NeighborhoodSelfAttention block for MI455X (gfx1250, wave32, WMMA).
// Pipeline: qkv GEMM -> 3D neighborhood attention -> proj GEMM -> instnorm ->
//           ffn1 GEMM(+gelu) -> ffn2 GEMM(+residual) -> instnorm (-> C,N).
// GEMMs: v_wmma_f32_16x16x32_bf16, 32x32 tile per wave (4 accumulators),
// bf16 tiles staged in LDS via packed v_cvt_pk_bf16_f32 converts.
// ---------------------------------------------------------------------------

#define NTOK 32768   // H*W*Z = 32^3
#define CDIM 128

typedef __attribute__((ext_vector_type(16))) __bf16 v16bf;
typedef __attribute__((ext_vector_type(8)))  __bf16 v8bf;
typedef __attribute__((ext_vector_type(2)))  __bf16 v2bf;
typedef __attribute__((ext_vector_type(8)))  float  v8f;
typedef __attribute__((ext_vector_type(2)))  float  v2f;

__device__ __forceinline__ unsigned short f2bf(float f) {
    __bf16 h = (__bf16)f;                       // hardware cvt on gfx1250
    return __builtin_bit_cast(unsigned short, h);
}

// Pair conversion -> single v_cvt_pk_bf16_f32, no repacking movs.
__device__ __forceinline__ unsigned int pk2bf(float x, float y) {
    v2f v = {x, y};
    v2bf b = __builtin_convertvector(v, v2bf);
    return __builtin_bit_cast(unsigned int, b);
}

__device__ __forceinline__ unsigned long long pk4bf(float4 v) {
    unsigned int lo = pk2bf(v.x, v.y);
    unsigned int hi = pk2bf(v.z, v.w);
    return (unsigned long long)lo | ((unsigned long long)hi << 32);
}

__device__ __forceinline__ float gelu_tanh(float x) {
    const float k0 = 0.7978845608028654f;  // sqrt(2/pi)
    const float k1 = 0.044715f;
    return 0.5f * x * (1.0f + tanhf(k0 * (x + k1 * x * x * x)));
}

// A fragment (16-bit A 16x32, ISA 7.12.2): lane half 0 -> K {0..7,16..23},
// half 1 -> K {8..15,24..31}; two 16B LDS reads per fragment.
__device__ __forceinline__ v16bf frag_a(const __bf16* row, unsigned half) {
    v8bf lo = *(const v8bf*)(row + half * 8);
    v8bf hi = *(const v8bf*)(row + 16 + half * 8);
    return __builtin_shufflevector(lo, hi,
               0,1,2,3,4,5,6,7,8,9,10,11,12,13,14,15);
}
// B fragment: lane holds column l16; same K split as A (half 0 -> K {0..7,16..23}, half 1 -> K {8..15,24..31}).
__device__ __forceinline__ v16bf frag_b(const __bf16* row, unsigned half) {
    v8bf lo = *(const v8bf*)(row + half * 8);
    v8bf hi = *(const v8bf*)(row + 16 + half * 8);
    return __builtin_shufflevector(lo, hi,
               0,1,2,3,4,5,6,7,8,9,10,11,12,13,14,15);
}

// ---------------------------------------------------------------------------
// Out[n][m] = epilogue( sum_k A[n][k] * W[m][k] + bias[m] )
//   A element (n,k) at: transA ? A[k*lda + n] : A[n*lda + k]
//   W is (Mout x Kdim) row-major.  act==1 -> gelu.  res != null -> +res[n][m].
// Block: 256 threads = 8 waves; block tile 128(N) x 64(M); K step = 32.
// Waves in 4x2 grid; each wave computes 32x32 = 2x2 WMMA tiles.
// All flat offsets fit in 32 bits -> unsigned index math only.
// ---------------------------------------------------------------------------
__global__ __launch_bounds__(256)
void gemm_bf16_wmma(const float* __restrict__ A, unsigned lda, int transA,
                    const float* __restrict__ W, const float* __restrict__ bias,
                    const float* __restrict__ res,
                    float* __restrict__ Out, unsigned Kdim, unsigned Mout, int act)
{
    // Row stride 40 halves = 80 B: keeps 16B alignment, staggers banks.
    __shared__ __align__(16) unsigned short As[128][40];
    __shared__ __align__(16) unsigned short Ws[64][40];

    const unsigned t    = threadIdx.x;
    const unsigned n0   = blockIdx.x * 128u;
    const unsigned j0   = blockIdx.y * 64u;
    const unsigned wave = t >> 5;
    const unsigned lane = t & 31u;
    const unsigned half = lane >> 4;
    const unsigned l16  = lane & 15u;
    const unsigned wr   = wave & 3u;    // 4 row blocks of 32
    const unsigned wc   = wave >> 2;    // 2 col blocks of 32

    v8f acc00 = {}, acc01 = {}, acc10 = {}, acc11 = {};

    const __bf16* arow0 = (const __bf16*)&As[wr * 32 + l16][0];
    const __bf16* arow1 = (const __bf16*)&As[wr * 32 + 16 + l16][0];
    const __bf16* brow0 = (const __bf16*)&Ws[wc * 32 + l16][0];
    const __bf16* brow1 = (const __bf16*)&Ws[wc * 32 + 16 + l16][0];

    #pragma unroll 2
    for (unsigned k0 = 0; k0 < Kdim; k0 += 32) {
        __syncthreads();
        if (transA) {
            // A stored (K, N): float4 along n (coalesced), scatter 4 rows.
            #pragma unroll
            for (int i = 0; i < 4; ++i) {
                unsigned idx = t + 256u * i;          // 1024 float4 tiles
                unsigned k = idx >> 5;
                unsigned r = (idx & 31u) * 4u;
                float4 v = *(const float4*)(A + ((k0 + k) * lda + n0 + r));
                As[r + 0][k] = f2bf(v.x);
                As[r + 1][k] = f2bf(v.y);
                As[r + 2][k] = f2bf(v.z);
                As[r + 3][k] = f2bf(v.w);
            }
        } else {
            // A stored (N, K): float4 along k, packed ds_store_b64.
            #pragma unroll
            for (int i = 0; i < 4; ++i) {
                unsigned idx = t + 256u * i;
                unsigned r = idx >> 3;
                unsigned k = (idx & 7u) * 4u;
                float4 v = *(const float4*)(A + ((n0 + r) * lda + k0 + k));
                *(unsigned long long*)&As[r][k] = pk4bf(v);
            }
        }
        #pragma unroll
        for (int i = 0; i < 2; ++i) {
            unsigned idx = t + 256u * i;              // 512 float4 tiles
            unsigned j = idx >> 3;
            unsigned k = (idx & 7u) * 4u;
            float4 v = *(const float4*)(W + ((j0 + j) * Kdim + k0 + k));
            *(unsigned long long*)&Ws[j][k] = pk4bf(v);
        }
        __syncthreads();

        v16bf a0 = frag_a(arow0, half);
        v16bf a1 = frag_a(arow1, half);
        v16bf b0 = frag_b(brow0, half);
        v16bf b1 = frag_b(brow1, half);

        acc00 = __builtin_amdgcn_wmma_f32_16x16x32_bf16(
                    false, a0, false, b0, (short)0, acc00, false, false);
        acc01 = __builtin_amdgcn_wmma_f32_16x16x32_bf16(
                    false, a0, false, b1, (short)0, acc01, false, false);
        acc10 = __builtin_amdgcn_wmma_f32_16x16x32_bf16(
                    false, a1, false, b0, (short)0, acc10, false, false);
        acc11 = __builtin_amdgcn_wmma_f32_16x16x32_bf16(
                    false, a1, false, b1, (short)0, acc11, false, false);
    }

    // D layout: VGPR r -> M = r + 8*half, N = l16.
    const unsigned gm0 = j0 + wc * 32 + l16;
    const unsigned gm1 = gm0 + 16;
    const float bj0 = bias[gm0];
    const float bj1 = bias[gm1];
    #pragma unroll
    for (int r = 0; r < 8; ++r) {
        unsigned gn0 = n0 + wr * 32 + r + 8 * half;
        unsigned gn1 = gn0 + 16;
        float v00 = acc00[r] + bj0, v01 = acc01[r] + bj1;
        float v10 = acc10[r] + bj0, v11 = acc11[r] + bj1;
        if (act == 1) {
            v00 = gelu_tanh(v00); v01 = gelu_tanh(v01);
            v10 = gelu_tanh(v10); v11 = gelu_tanh(v11);
        }
        if (res) {
            v00 += res[gn0 * Mout + gm0]; v01 += res[gn0 * Mout + gm1];
            v10 += res[gn1 * Mout + gm0]; v11 += res[gn1 * Mout + gm1];
        }
        Out[gn0 * Mout + gm0] = v00;
        Out[gn0 * Mout + gm1] = v01;
        Out[gn1 * Mout + gm0] = v10;
        Out[gn1 * Mout + gm1] = v11;
    }
}

// ---------------------------------------------------------------------------
// 3D neighborhood attention, K=3 (27 neighbors), one thread per (token, head).
// qkv: (N, 384) with [q|k|v] each 128 = 8 heads x 16.  rpb: (8,5,5,5).
// out: (N, 128).
// ---------------------------------------------------------------------------
__global__ __launch_bounds__(256)
void na3d_attn(const float* __restrict__ qkv, const float* __restrict__ rpb,
               float* __restrict__ out)
{
    unsigned t = blockIdx.x * 256u + threadIdx.x;
    unsigned n = t >> 3;
    unsigned h = t & 7u;
    int z = n & 31, w = (n >> 5) & 31, x = n >> 10;
    int sx = min(max(x - 1, 0), 29);
    int sy = min(max(w - 1, 0), 29);
    int sz = min(max(z - 1, 0), 29);

    const float scale = 0.25f;  // 1/sqrt(16)
    const float* qp = qkv + n * 384u + h * 16u;
    float q[16];
    #pragma unroll
    for (int d = 0; d < 16; ++d) q[d] = qp[d] * scale;

    float logits[27];
    float mx = -1e30f;
    #pragma unroll
    for (int a = 0; a < 3; ++a)
      for (int b = 0; b < 3; ++b)
        for (int c = 0; c < 3; ++c) {
            unsigned nn = (unsigned)((((sx + a) * 32) + (sy + b)) * 32 + (sz + c));
            const float* kp = qkv + nn * 384u + 128u + h * 16u;
            float acc = 0.f;
            #pragma unroll
            for (int d = 0; d < 16; ++d) acc += q[d] * kp[d];
            int rh = sx + a - x + 2, rw = sy + b - w + 2, rz = sz + c - z + 2;
            acc += rpb[(((h * 5) + rh) * 5 + rw) * 5 + rz];
            logits[(a * 3 + b) * 3 + c] = acc;
            mx = fmaxf(mx, acc);
        }

    float outv[16];
    #pragma unroll
    for (int d = 0; d < 16; ++d) outv[d] = 0.f;
    float den = 0.f;
    #pragma unroll
    for (int a = 0; a < 3; ++a)
      for (int b = 0; b < 3; ++b)
        for (int c = 0; c < 3; ++c) {
            float p = __expf(logits[(a * 3 + b) * 3 + c] - mx);
            den += p;
            unsigned nn = (unsigned)((((sx + a) * 32) + (sy + b)) * 32 + (sz + c));
            const float* vp = qkv + nn * 384u + 256u + h * 16u;
            #pragma unroll
            for (int d = 0; d < 16; ++d) outv[d] += p * vp[d];
        }
    float inv = 1.f / den;
    float* op = out + n * 128u + h * 16u;
    #pragma unroll
    for (int d = 0; d < 16; ++d) op[d] = outv[d] * inv;
}

// ---------------------------------------------------------------------------
// Instance norm over N tokens per channel.  in: (N, C).
// toCN==0: out (N, C);  toCN==1: out (C, N)  (final layout (B,C,H,W,Z)).
// One block per channel, 256 threads, LDS tree reduction.
// ---------------------------------------------------------------------------
__global__ __launch_bounds__(256)
void inst_norm(const float* __restrict__ in, float* __restrict__ out, int toCN)
{
    __shared__ float s1[256], s2[256];
    const unsigned c = blockIdx.x;
    const unsigned t = threadIdx.x;
    float sum = 0.f, sq = 0.f;
    for (unsigned n = t; n < NTOK; n += 256u) {
        float v = in[n * CDIM + c];
        sum += v; sq += v * v;
    }
    s1[t] = sum; s2[t] = sq;
    __syncthreads();
    for (unsigned s = 128; s > 0; s >>= 1) {
        if (t < s) { s1[t] += s1[t + s]; s2[t] += s2[t + s]; }
        __syncthreads();
    }
    float mean = s1[0] * (1.f / NTOK);
    float var  = s2[0] * (1.f / NTOK) - mean * mean;
    float rstd = rsqrtf(var + 1e-5f);
    for (unsigned n = t; n < NTOK; n += 256u) {
        float v = (in[n * CDIM + c] - mean) * rstd;
        if (toCN) out[c * NTOK + n] = v;
        else      out[n * CDIM + c] = v;
    }
}

// ---------------------------------------------------------------------------
extern "C" void kernel_launch(void* const* d_in, const int* in_sizes, int n_in,
                              void* d_out, int out_size, void* d_ws, size_t ws_size,
                              hipStream_t stream)
{
    (void)in_sizes; (void)n_in; (void)out_size; (void)ws_size;

    const float* x      = (const float*)d_in[0];  // (C, N)
    const float* w_qkv  = (const float*)d_in[1];  // (384, 128)
    const float* b_qkv  = (const float*)d_in[2];
    const float* rpb    = (const float*)d_in[3];  // (8,5,5,5)
    const float* w_proj = (const float*)d_in[4];  // (128, 128)
    const float* b_proj = (const float*)d_in[5];
    const float* w_ffn1 = (const float*)d_in[6];  // (512, 128)
    const float* b_ffn1 = (const float*)d_in[7];
    const float* w_ffn2 = (const float*)d_in[8];  // (128, 512)
    const float* b_ffn2 = (const float*)d_in[9];
    float* out = (float*)d_out;                   // (C, N)

    // Workspace partition (floats), buffers reused once dead:
    //   bufA: qkv (N*384) -> proj_out -> t2 (N*128)
    //   bufB: attn_out -> norm1 (N*128)
    //   bufC: ffn1 (N*512)
    float* ws   = (float*)d_ws;
    const size_t NQKV = (size_t)NTOK * 384;
    const size_t NC   = (size_t)NTOK * 128;
    float* qkv   = ws;
    float* attn  = ws + NQKV;
    float* ffn1  = ws + NQKV + NC;
    float* proj  = qkv;
    float* norm1 = attn;
    float* t2    = qkv;

    dim3 blk(256);

    // 1) qkv = x^T @ w_qkv^T + b_qkv          (A = x transposed view)
    gemm_bf16_wmma<<<dim3(NTOK / 128, 384 / 64), blk, 0, stream>>>(
        x, NTOK, 1, w_qkv, b_qkv, nullptr, qkv, 128, 384, 0);

    // 2) neighborhood attention -> (N, C)
    na3d_attn<<<dim3((NTOK * 8) / 256), blk, 0, stream>>>(qkv, rpb, attn);

    // 3) proj = attn @ w_proj^T + b_proj
    gemm_bf16_wmma<<<dim3(NTOK / 128, 128 / 64), blk, 0, stream>>>(
        attn, 128, 0, w_proj, b_proj, nullptr, proj, 128, 128, 0);

    // 4) instance norm -> norm1 (N, C)
    inst_norm<<<dim3(128), blk, 0, stream>>>(proj, norm1, 0);

    // 5) ffn1 = gelu(norm1 @ w_ffn1^T + b_ffn1)    (N, 512)
    gemm_bf16_wmma<<<dim3(NTOK / 128, 512 / 64), blk, 0, stream>>>(
        norm1, 128, 0, w_ffn1, b_ffn1, nullptr, ffn1, 128, 512, 1);

    // 6) t2 = norm1 + ffn1 @ w_ffn2^T + b_ffn2     (N, 128)
    gemm_bf16_wmma<<<dim3(NTOK / 128, 128 / 64), blk, 0, stream>>>(
        ffn1, 512, 0, w_ffn2, b_ffn2, norm1, t2, 512, 128, 0);

    // 7) final instance norm, written transposed to (C, N)
    inst_norm<<<dim3(128), blk, 0, stream>>>(t2, out, 1);
}
